// StaticInt8Linear_40286793236698
// MI455X (gfx1250) — 8-bit matrix operand order repaired and checked by reading, compiled, not run on hardware
//
#include <hip/hip_runtime.h>
#include <stdint.h>

typedef __attribute__((ext_vector_type(8))) int   v8i;
typedef __attribute__((ext_vector_type(4))) float f32x4;

#define M_DIM 8192
#define N_DIM 16384
#define K_DIM 4096

#define BM 128
#define BN 128
#define BK 128
#define LDS_STRIDE 144
#define NSTAGE (K_DIM / BK)

__device__ __forceinline__ void async_copy_b128(unsigned lds_off, const void* gptr) {
    uint64_t ga = (uint64_t)(uintptr_t)gptr;
    asm volatile("global_load_async_to_lds_b128 %0, %1, off"
                 :: "v"(lds_off), "v"(ga)
                 : "memory");
}
__device__ __forceinline__ void wait_asynccnt0() {
    asm volatile("s_wait_asynccnt 0x0" ::: "memory");
}

__global__ __launch_bounds__(256) void quantize_act(const float* __restrict__ x,
                                                    const float* __restrict__ act_scale,
                                                    int8_t* __restrict__ xq,
                                                    int total4) {
    int idx = blockIdx.x * blockDim.x + threadIdx.x;
    if (idx >= total4) return;
    const float inv = 1.0f / act_scale[0];
    const f32x4 v = __builtin_nontemporal_load(((const f32x4*)x) + idx);
    int q0 = __float2int_rn(v.x * inv);
    int q1 = __float2int_rn(v.y * inv);
    int q2 = __float2int_rn(v.z * inv);
    int q3 = __float2int_rn(v.w * inv);
    q0 = max(-128, min(127, q0));
    q1 = max(-128, min(127, q1));
    q2 = max(-128, min(127, q2));
    q3 = max(-128, min(127, q3));
    int packed = (q0 & 255) | ((q1 & 255) << 8) | ((q2 & 255) << 16) | ((q3 & 255) << 24);
    ((int*)xq)[idx] = packed;
}

__global__ __launch_bounds__(256) void int8_gemm_wmma(const int8_t* __restrict__ A,
                                                      const int8_t* __restrict__ W,
                                                      const float* __restrict__ wscale,
                                                      const float* __restrict__ act_scale,
                                                      const float* __restrict__ bias,
                                                      float* __restrict__ out) {
    __shared__ int8_t As[2][BM * LDS_STRIDE];
    __shared__ int8_t Bs[2][BN * LDS_STRIDE];

    const int tid   = threadIdx.x;
    const int lane  = tid & 31;
    const int wid   = tid >> 5;
    const int waveM = wid & 3;
    const int waveN = wid >> 2;
    const int h     = lane >> 4;
    const int lm    = lane & 15;

    const int blockN0 = blockIdx.x * BN;
    const int blockM0 = blockIdx.y * BM;

    const v8i vzero = {0, 0, 0, 0, 0, 0, 0, 0};
    v8i acc[2][4];
#pragma unroll
    for (int mt = 0; mt < 2; ++mt)
#pragma unroll
        for (int nt = 0; nt < 4; ++nt) acc[mt][nt] = vzero;

    int srow[4], scol[4];
#pragma unroll
    for (int i = 0; i < 4; ++i) {
        const int chunk = tid + i * 256;
        srow[i] = chunk >> 3;
        scol[i] = (chunk & 7) * 16;
    }

#pragma unroll
    for (int i = 0; i < 4; ++i) {
        async_copy_b128((unsigned)(uintptr_t)(As[0] + srow[i] * LDS_STRIDE + scol[i]),
                        A + (size_t)(blockM0 + srow[i]) * K_DIM + scol[i]);
        async_copy_b128((unsigned)(uintptr_t)(Bs[0] + srow[i] * LDS_STRIDE + scol[i]),
                        W + (size_t)(blockN0 + srow[i]) * K_DIM + scol[i]);
    }

    for (int s = 0; s < NSTAGE; ++s) {
        const int cur = s & 1;
        const int nxt = cur ^ 1;

        wait_asynccnt0();
        __syncthreads();

        if (s + 1 < NSTAGE) {
            const int k0n = (s + 1) * BK;
#pragma unroll
            for (int i = 0; i < 4; ++i) {
                async_copy_b128((unsigned)(uintptr_t)(As[nxt] + srow[i] * LDS_STRIDE + scol[i]),
                                A + (size_t)(blockM0 + srow[i]) * K_DIM + k0n + scol[i]);
                async_copy_b128((unsigned)(uintptr_t)(Bs[nxt] + srow[i] * LDS_STRIDE + scol[i]),
                                W + (size_t)(blockN0 + srow[i]) * K_DIM + k0n + scol[i]);
            }
        }

        const int8_t* curA = As[cur];
        const int8_t* curB = Bs[cur];

#pragma unroll
        for (int kk = 0; kk < 2; ++kk) {
            const int ko = kk * 64;

            v8i afrag[2];
#pragma unroll
            for (int mt = 0; mt < 2; ++mt) {
                const int8_t* base = curA + (waveM * 32 + mt * 16 + lm) * LDS_STRIDE + ko;
#pragma unroll
                for (int c = 0; c < 4; ++c) {
                    const int* p = (const int*)(base + (c >> 1) * 32 + h * 16 + (c & 1) * 8);
                    afrag[mt][2 * c]     = p[0];
                    afrag[mt][2 * c + 1] = p[1];
                }
            }

            v8i bfrag[4];
#pragma unroll
            for (int nt = 0; nt < 4; ++nt) {
                const int8_t* base = curB + (waveN * 64 + nt * 16 + lm) * LDS_STRIDE + ko + h * 16;
                const int4 b0 = *(const int4*)(base);
                const int4 b1 = *(const int4*)(base + 32);
                bfrag[nt][0] = b0.x; bfrag[nt][1] = b0.y;
                bfrag[nt][2] = b0.z; bfrag[nt][3] = b0.w;
                bfrag[nt][4] = b1.x; bfrag[nt][5] = b1.y;
                bfrag[nt][6] = b1.z; bfrag[nt][7] = b1.w;
            }

#pragma unroll
            for (int mt = 0; mt < 2; ++mt)
#pragma unroll
                for (int nt = 0; nt < 4; ++nt)
                    acc[mt][nt] = __builtin_amdgcn_wmma_i32_16x16x64_iu8(
                         true, afrag[mt],
                         true, bfrag[nt],
                        acc[mt][nt],
                         false,  false);
        }
    }

    const float as = act_scale[0];
#pragma unroll
    for (int nt = 0; nt < 4; ++nt) {
        const int gcol = blockN0 + waveN * 64 + nt * 16 + lm;
        const float os = as * wscale[gcol];
        const float bv = bias[gcol];
#pragma unroll
        for (int mt = 0; mt < 2; ++mt) {
            const int growBase = blockM0 + waveM * 32 + mt * 16 + h * 8;
#pragma unroll
            for (int r = 0; r < 8; ++r) {
                __builtin_nontemporal_store(
                    (float)acc[mt][nt][r] * os + bv,
                    out + (size_t)(growBase + r) * N_DIM + gcol);
            }
        }
    }
}

extern "C" void kernel_launch(void* const* d_in, const int* in_sizes, int n_in,
                              void* d_out, int out_size, void* d_ws, size_t ws_size,
                              hipStream_t stream) {
    const float*  x      = (const float*)d_in[0];
    const int8_t* weight = (const int8_t*)d_in[1];
    const float*  wscale = (const float*)d_in[2];
    const float*  ascale = (const float*)d_in[3];
    const float*  bias   = (const float*)d_in[4];
    float*        out    = (float*)d_out;
    int8_t*       xq     = (int8_t*)d_ws;

    const int total4 = (M_DIM * K_DIM) / 4;
    quantize_act<<<total4 / 256, 256, 0, stream>>>(x, ascale, xq, total4);

    dim3 grid(N_DIM / BN, M_DIM / BM);
    int8_gemm_wmma<<<grid, 256, 0, stream>>>(xq, weight, wscale, ascale, bias, out);
}
